// MultiHeadAttention_30090540876366
// MI455X (gfx1250) — matrix operand order repaired and checked by reading, not compiled, not run on hardware
//
#include <hip/hip_runtime.h>
#include <hip/hip_bf16.h>

// ---------------------------------------------------------------------------
// MHA forward for MI455X (gfx1250), wave32 + WMMA f16->f32.
//   B=2, S=2048, D=1024, H=16, K=64
// Pipeline: f32->f16 convert, weight transpose, 3 WMMA GEMMs (Q,K; V stored
// transposed per batch), flash-style WMMA attention with cooperative
// double-buffered LDS staging of K/V tiles, WMMA output projection.
// ---------------------------------------------------------------------------

#define B_ 2
#define S_ 2048
#define D_ 1024
#define H_ 16
#define KH_ 64

#define KPAD 72   // 32x64 K tile rows padded: 144B row = 9x16B, bank step 36
#define VPAD 40   // 64x32 V tile rows padded:  80B row = 5x16B, bank step 20

typedef _Float16 half16 __attribute__((ext_vector_type(16)));
typedef _Float16 half8  __attribute__((ext_vector_type(8)));
typedef _Float16 half4v __attribute__((ext_vector_type(4)));
typedef float    float8 __attribute__((ext_vector_type(8)));

__device__ __forceinline__ float8 wmma16x16x32(half16 a, half16 b, float8 c) {
    // D = A(16x32 f16) * B(32x16 f16) + C(16x16 f32)
    return __builtin_amdgcn_wmma_f32_16x16x32_f16(
        /*neg_a=*/false, a, /*neg_b=*/false, b,
        /*c_mod=*/(short)0, c, /*reuse_a=*/false, /*reuse_b=*/false);
}

__device__ __forceinline__ half16 cat8(half8 lo, half8 hi) {
    return __builtin_shufflevector(lo, hi, 0, 1, 2, 3, 4, 5, 6, 7,
                                   8, 9, 10, 11, 12, 13, 14, 15);
}

// A-matrix fragment, K=32, 16-bit (ISA 7.12.2):
//  frag[0..7] = K(group*8..+7), frag[8..15] = K(16+group*8..+7): two b128.
template <typename PT>
__device__ __forceinline__ half16 ldfragA(const PT* p, int group) {
    half8 lo = *(const half8*)(p + group * 8);
    half8 hi = *(const half8*)(p + 16 + group * 8);
    return cat8(lo, hi);
}

// B-matrix fragment, K=32, 16-bit (same k order as A):
//  frag[0..7] = K(group*8..+7), frag[8..15] = K(16+group*8..+7): two b128.
template <typename PT>
__device__ __forceinline__ half16 ldfragB(const PT* p, int group) {
    half8 lo = *(const half8*)(p + group * 8);
    half8 hi = *(const half8*)(p + 16 + group * 8);
    return cat8(lo, hi);
}

// ---------------------------------------------------------------------------
// f32 -> f16 elementwise convert (x4 vectorized)
// ---------------------------------------------------------------------------
__global__ void cvt_f32_f16_x4(const float* __restrict__ in,
                               _Float16* __restrict__ out, int n4) {
    int i = blockIdx.x * 256 + threadIdx.x;
    if (i < n4) {
        float4 v = ((const float4*)in)[i];
        half4v o = {(_Float16)v.x, (_Float16)v.y, (_Float16)v.z, (_Float16)v.w};
        ((half4v*)out)[i] = o;
    }
}

// ---------------------------------------------------------------------------
// Weight transpose + convert: Wt[n*Dd + k] = (f16) W[k*Dd + n]
// ---------------------------------------------------------------------------
__global__ void transpose_cvt(const float* __restrict__ W,
                              _Float16* __restrict__ Wt, int Dd) {
    int idx = blockIdx.x * 256 + threadIdx.x;
    if (idx < Dd * Dd) {
        int n = idx / Dd;
        int k = idx - n * Dd;
        Wt[(size_t)n * Dd + k] = (_Float16)W[(size_t)k * Dd + n];
    }
}

// ---------------------------------------------------------------------------
// GEMM: Out = A[M,Kd] @ Bt[N,Kd]^T + bias[N].  One wave per 16x64 tile
// (4 accumulators, A-fragment reused 4x -> 4 WMMAs per K-step).
// MODE 0: f16 row-major [M,N]
// MODE 1: f32 row-major [M,N]
// MODE 2: f16 transposed per batch: Out[b][n][s] with M = B_*S_ (for V)
// ---------------------------------------------------------------------------
template <int MODE>
__global__ __launch_bounds__(256) void gemm_wmma(
    const _Float16* __restrict__ A, const _Float16* __restrict__ Bt,
    const float* __restrict__ bias, void* __restrict__ Out,
    int M, int N, int Kd) {
    const int lane  = threadIdx.x & 31;
    const int wave  = threadIdx.x >> 5;
    const int gw    = blockIdx.x * 8 + wave;
    const int tilesN = N >> 6;                  // 64-wide wave tiles
    const int tm = gw / tilesN;
    const int tn = gw - tm * tilesN;
    if (tm * 16 >= M) return;                   // uniform per wave

    const int group = lane >> 4;
    const int nl    = lane & 15;
    const _Float16* arow = A + (size_t)(tm * 16 + nl) * Kd;
    const _Float16* brow[4];
#pragma unroll
    for (int c = 0; c < 4; ++c)
        brow[c] = Bt + (size_t)(tn * 64 + c * 16 + nl) * Kd;

    float8 acc[4] = {};
    for (int k0 = 0; k0 < Kd; k0 += 32) {
        // hoist all loads ahead of the WMMAs so they can issue as one clause
        half16 af = ldfragA(arow + k0, group);
        half16 bf[4];
#pragma unroll
        for (int c = 0; c < 4; ++c) bf[c] = ldfragB(brow[c] + k0, group);
#pragma unroll
        for (int c = 0; c < 4; ++c) acc[c] = wmma16x16x32(af, bf[c], acc[c]);
    }

#pragma unroll
    for (int c = 0; c < 4; ++c) {
        const int col = tn * 64 + c * 16 + nl;
        const float bb = bias[col];
        if (MODE == 2) {
            // column of 8 consecutive rows -> 8 consecutive s -> one b128
            const int row0 = tm * 16 + group * 8;
            const int bidx = row0 >> 11;        // row0 / S_
            const int s0   = row0 & (S_ - 1);
            half8 o;
#pragma unroll
            for (int r = 0; r < 8; ++r) o[r] = (_Float16)(acc[c][r] + bb);
            *(half8*)((_Float16*)Out + ((size_t)bidx * N + col) * S_ + s0) = o;
        } else {
#pragma unroll
            for (int r = 0; r < 8; ++r) {
                size_t row = (size_t)(tm * 16 + group * 8 + r);
                float v = acc[c][r] + bb;
                if (MODE == 1) ((float*)Out)[row * N + col] = v;
                else           ((_Float16*)Out)[row * N + col] = (_Float16)v;
            }
        }
    }
}

// ---------------------------------------------------------------------------
// Flash-style attention.  Block = 8 waves, all on the SAME (b,h) with 8
// consecutive 16-row q tiles, so K/V tiles are staged cooperatively in LDS
// once per 32-key step (double buffered, one barrier per step).
// Per step & wave: 4 WMMAs for Q*K^T, online softmax with the reference's
// multiplicative mask (score *= mask; score==0 -> -1e9), P re-layout through
// padded LDS, 4 WMMAs for P*V.
// ---------------------------------------------------------------------------
__global__ __launch_bounds__(256) void attn_wmma(
    const _Float16* __restrict__ Qf, const _Float16* __restrict__ Kf,
    const _Float16* __restrict__ Vt, const float* __restrict__ mask,
    _Float16* __restrict__ ctx) {
    __shared__ __align__(16) _Float16 Ksh[2][32][KPAD];   // keys x d
    __shared__ __align__(16) _Float16 Vsh[2][64][VPAD];   // d x keys
    __shared__ __align__(16) _Float16 Psh[8][16][VPAD];   // per-wave P staging

    const int tid  = threadIdx.x;
    const int lane = tid & 31;
    const int wave = tid >> 5;
    // block -> (b, h, group of 8 q-tiles)
    const int qg = blockIdx.x & (S_ / 128 - 1);          // 16 groups
    const int hh = (blockIdx.x / (S_ / 128)) % H_;
    const int b  = blockIdx.x / ((S_ / 128) * H_);
    const int group = lane >> 4;
    const int nl    = lane & 15;
    const int q0    = (qg * 8 + wave) * 16;

    // ---- cooperative staging source pointers (per thread: one b128 each) ---
    const int krow  = tid >> 3;                // 0..31 (key within tile)
    const int kcol8 = (tid & 7) * 8;           // d offset
    const _Float16* kgsrc =
        Kf + (size_t)(b * S_ + krow) * D_ + hh * KH_ + kcol8;
    const int vrow  = tid >> 2;                // 0..63 (d within head)
    const int vcol8 = (tid & 3) * 8;           // key offset
    const _Float16* vhead = Vt + ((size_t)b * D_ + hh * KH_) * S_;
    const _Float16* vgsrc = vhead + (size_t)vrow * S_ + vcol8;

    // ---- load Q A-fragments (16 rows x 64 d -> two K=32 fragments) ----
    const _Float16* qp = Qf + (size_t)(b * S_ + q0 + nl) * D_ + hh * KH_;
    half16 qa0 = ldfragA(qp, group);
    half16 qa1 = ldfragA(qp + 32, group);

    float m[8], l[8];
    float8 acc[4] = {};
#pragma unroll
    for (int r = 0; r < 8; ++r) { m[r] = -__builtin_inff(); l[r] = 0.0f; }

    const float scale = 0.125f;               // 1/sqrt(64)
    _Float16* P = &Psh[wave][0][0];

    // prime buffer 0
    *(half8*)&Ksh[0][krow][kcol8] = *(const half8*)(kgsrc);
    *(half8*)&Vsh[0][vrow][vcol8] = *(const half8*)(vgsrc);
    __syncthreads();

    const int NIT = S_ / 32;                  // 64 steps
    for (int it = 0; it < NIT; ++it) {
        const int key0 = it * 32;
        const int cur  = it & 1;
        // stage next tile into the other buffer (overlaps compute)
        if (it + 1 < NIT) {
            *(half8*)&Ksh[cur ^ 1][krow][kcol8] =
                *(const half8*)(kgsrc + (size_t)(key0 + 32) * D_);
            *(half8*)&Vsh[cur ^ 1][vrow][vcol8] =
                *(const half8*)(vgsrc + key0 + 32);
        }
        // prefetch two tiles ahead (gfx1250 global_prefetch_b8)
        if (it + 2 < NIT) {
            __builtin_prefetch(kgsrc + (size_t)(key0 + 64) * D_, 0, 0);
            __builtin_prefetch(vgsrc + key0 + 64, 0, 0);
        }

        // ---- scores: hoist 4 K-fragments (8 ds_load_b128), then 4 WMMAs
        //      interleaved across the two independent accumulators ----
        const _Float16* kp0 = &Ksh[cur][nl][0];
        const _Float16* kp1 = &Ksh[cur][16 + nl][0];
        half16 kb0 = ldfragB(kp0, group);
        half16 kb1 = ldfragB(kp0 + 32, group);
        half16 kb2 = ldfragB(kp1, group);
        half16 kb3 = ldfragB(kp1 + 32, group);
        float8 sc[2] = {{}, {}};
        sc[0] = wmma16x16x32(qa0, kb0, sc[0]);
        sc[1] = wmma16x16x32(qa0, kb2, sc[1]);
        sc[0] = wmma16x16x32(qa1, kb1, sc[0]);
        sc[1] = wmma16x16x32(qa1, kb3, sc[1]);

        // ---- reference mask semantics + online softmax update ----
#pragma unroll
        for (int r = 0; r < 8; ++r) {
            const int qrow = q0 + group * 8 + r;
            const size_t mbase = ((size_t)b * S_ + qrow) * S_;
#pragma unroll
            for (int t = 0; t < 2; ++t) {
                int kcol = key0 + t * 16 + nl;
                float mv = mask[mbase + kcol];
                float sv = sc[t][r] * scale * mv;
                sc[t][r] = (sv == 0.0f) ? -1e9f : sv;
            }
            float rm = fmaxf(sc[0][r], sc[1][r]);
#pragma unroll
            for (int x = 1; x <= 8; x <<= 1)
                rm = fmaxf(rm, __shfl_xor(rm, x, 32));   // 16-lane half reduce
            float mn   = fmaxf(m[r], rm);
            float corr = __expf(m[r] - mn);
            m[r] = mn;
            float p0 = __expf(sc[0][r] - mn);
            float p1 = __expf(sc[1][r] - mn);
            float rs = p0 + p1;
#pragma unroll
            for (int x = 1; x <= 8; x <<= 1)
                rs += __shfl_xor(rs, x, 32);
            l[r] = l[r] * corr + rs;
#pragma unroll
            for (int c = 0; c < 4; ++c) acc[c][r] *= corr;
            // C-layout -> padded LDS rows
            P[(group * 8 + r) * VPAD + nl]      = (_Float16)p0;
            P[(group * 8 + r) * VPAD + 16 + nl] = (_Float16)p1;
        }

        // ---- re-load P as A-fragment (wave-private LDS) ----
        half16 pa = ldfragA(P + nl * VPAD, group);

        // ---- P(16x32) @ V(32x64): hoist 4 V-fragments, then 4 WMMAs ----
        half16 vb[4];
#pragma unroll
        for (int c = 0; c < 4; ++c)
            vb[c] = ldfragB(&Vsh[cur][c * 16 + nl][0], group);
#pragma unroll
        for (int c = 0; c < 4; ++c)
            acc[c] = wmma16x16x32(pa, vb[c], acc[c]);

        __syncthreads();   // staging of buf^1 done; reads of buf done
    }

    // ---- normalize and store ctx (f16, heads packed along D) ----
#pragma unroll
    for (int r = 0; r < 8; ++r) {
        float inv = 1.0f / l[r];
        size_t row = (size_t)(b * S_ + q0 + group * 8 + r);
#pragma unroll
        for (int c = 0; c < 4; ++c)
            ctx[row * D_ + hh * KH_ + c * 16 + nl] =
                (_Float16)(acc[c][r] * inv);
    }
}

// ---------------------------------------------------------------------------
// Host-side launch
// ---------------------------------------------------------------------------
extern "C" void kernel_launch(void* const* d_in, const int* in_sizes, int n_in,
                              void* d_out, int out_size, void* d_ws, size_t ws_size,
                              hipStream_t stream) {
    const float* dec  = (const float*)d_in[0];
    const float* enc  = (const float*)d_in[1];
    const float* mask = (const float*)d_in[2];
    const float* Wq = (const float*)d_in[3];  const float* bq = (const float*)d_in[4];
    const float* Wk = (const float*)d_in[5];  const float* bk = (const float*)d_in[6];
    const float* Wv = (const float*)d_in[7];  const float* bv = (const float*)d_in[8];
    const float* Wo = (const float*)d_in[9];  const float* bo = (const float*)d_in[10];
    float* out = (float*)d_out;

    const size_t BSD = (size_t)B_ * S_ * D_;   // 4,194,304
    const size_t DD  = (size_t)D_ * D_;        // 1,048,576

    char* ws = (char*)d_ws;
    size_t off = 0;
    auto halloc = [&](size_t nhalf) -> _Float16* {
        _Float16* p = (_Float16*)(ws + off);
        off = (off + nhalf * sizeof(_Float16) + 255) & ~(size_t)255;
        return p;
    };
    _Float16* hdec = halloc(BSD);
    _Float16* henc = halloc(BSD);
    _Float16* WqT  = halloc(DD);
    _Float16* WkT  = halloc(DD);
    _Float16* WvT  = halloc(DD);
    _Float16* WoT  = halloc(DD);
    _Float16* Qh   = halloc(BSD);
    _Float16* Kh   = halloc(BSD);
    _Float16* VhT  = halloc(BSD);   // [B][D][S]
    _Float16* Ch   = halloc(BSD);
    (void)ws_size; (void)in_sizes; (void)n_in; (void)out_size;

    // 1) activation conversion
    {
        int n4 = (int)(BSD / 4);
        int blocks = (n4 + 255) / 256;
        cvt_f32_f16_x4<<<blocks, 256, 0, stream>>>(dec, hdec, n4);
        cvt_f32_f16_x4<<<blocks, 256, 0, stream>>>(enc, henc, n4);
    }
    // 2) weight transpose+convert
    {
        int blocks = (int)((DD + 255) / 256);
        transpose_cvt<<<blocks, 256, 0, stream>>>(Wq, WqT, D_);
        transpose_cvt<<<blocks, 256, 0, stream>>>(Wk, WkT, D_);
        transpose_cvt<<<blocks, 256, 0, stream>>>(Wv, WvT, D_);
        transpose_cvt<<<blocks, 256, 0, stream>>>(Wo, WoT, D_);
    }
    // 3) Q/K/V projections: M = B*S = 4096, N = Kd = 1024, wave tile 16x64
    {
        const int M = B_ * S_, N = D_, Kd = D_;
        int blocks = (M / 16) * (N / 64) / 8;   // 512
        gemm_wmma<0><<<blocks, 256, 0, stream>>>(hdec, WqT, bq, Qh,  M, N, Kd);
        gemm_wmma<0><<<blocks, 256, 0, stream>>>(henc, WkT, bk, Kh,  M, N, Kd);
        gemm_wmma<2><<<blocks, 256, 0, stream>>>(henc, WvT, bv, VhT, M, N, Kd);
    }
    // 4) attention: 512 blocks, each = one (b,h) x 128 q rows
    {
        int blocks = B_ * H_ * (S_ / 128);      // 512
        attn_wmma<<<blocks, 256, 0, stream>>>(Qh, Kh, VhT, mask, Ch);
    }
    // 5) output projection -> f32 result
    {
        const int M = B_ * S_, N = D_, Kd = D_;
        int blocks = (M / 16) * (N / 64) / 8;
        gemm_wmma<1><<<blocks, 256, 0, stream>>>(Ch, WoT, bo, out, M, N, Kd);
    }
}
